// AGGCN_23364622090652
// MI455X (gfx1250) — matrix operand order repaired and checked by reading, not compiled, not run on hardware
//
#include <hip/hip_runtime.h>
#include <math.h>

#define B_  16
#define S_  64
#define N_  4096
#define E_  32768
#define ET_ 36864            // E_ + N_ self loops
#define BN_ (B_ * N_)        // 65536
#define NCOL_ (B_ * 64)      // 1024, fused (b,d) column dim of the big GEMM

typedef unsigned short u16;
typedef __attribute__((ext_vector_type(16))) __bf16 bf16x16;
typedef __attribute__((ext_vector_type(8)))  float  f32x8;
typedef __attribute__((ext_vector_type(8)))  short  s16x8;
typedef __attribute__((ext_vector_type(4)))  unsigned int u32x4;
typedef __attribute__((ext_vector_type(8)))  int    i32x8;
typedef __attribute__((ext_vector_type(4)))  int    i32x4;

union FragBF { bf16x16 bf; s16x8 h[2]; };

__device__ __forceinline__ u16 f2bf(float x) {
  unsigned u = __float_as_uint(x);
  unsigned r = (u + 0x7FFFu + ((u >> 16) & 1u)) >> 16;   // round-to-nearest-even
  return (u16)r;
}

__device__ __forceinline__ void atomicMaxF(float* addr, float v) {
  if (v >= 0.f) atomicMax((int*)addr, __float_as_int(v));
  else          atomicMin((unsigned int*)addr, (unsigned int)__float_as_int(v));
}

// ---------------------------------------------------------------------------
// TDM: async-copy a 128-row x 32-col tile of 16-bit data (row stride = K
// elements) from global into LDS (packed row-major, 8 KB). D# built per ISA
// CDNA5 ISA: count=1, type=2, data_size=2B, tile 32x128,
// tensor dims saturated (tiles never go OOB), no padding/iteration/multicast.
// 6-arg builtin form (clang-23 / therock lane).
// ---------------------------------------------------------------------------
__device__ __forceinline__ void tdm_load_tile_128x32(const u16* gptr,
                                                     unsigned lds_off, int K) {
  unsigned long long ga = (unsigned long long)(uintptr_t)gptr;
  u32x4 g0;
  g0[0] = 1u;                                          // count=1 (valid)
  g0[1] = lds_off;                                     // lds_addr
  g0[2] = (unsigned)ga;                                // global_addr[31:0]
  g0[3] = (unsigned)((ga >> 32) & 0x01FFFFFFu) | (2u << 30);  // addr[56:32]|type=2
  i32x8 g1;
  g1[0] = 0x00010000;                                  // data_size = 2 bytes
  g1[1] = (int)0xFFFF0000u;                            // tensor_dim0 lo16<<16
  g1[2] = (int)0xFFFF7FFFu;                            // dim0 hi16 | dim1 lo16
  g1[3] = (int)((32u << 16) | 0x7FFFu);                // dim1 hi16 | tile_dim0=32
  g1[4] = 128;                                         // tile_dim1=128, tile_dim2=0
  g1[5] = K;                                           // tensor_dim0_stride (elems)
  g1[6] = 0;
  g1[7] = 0;
  i32x4 z4 = {0, 0, 0, 0};
  i32x8 z8 = {0, 0, 0, 0, 0, 0, 0, 0};
  __builtin_amdgcn_tensor_load_to_lds(g0, g1, z4, z4, z8, 0);
}

// ---------------------------------------------------------------------------
// bf16 GEMM, f32 accumulate: C[M,N] = A[M,K] * BT[N,K]^T
// 256 threads = 8 waves, tile 128x128, BK=32, v_wmma_f32_16x16x32_bf16.
// Both LDS tiles ([m][k] and [n][k]) are straight 2D copies -> streamed by
// the Tensor Data Mover, double-buffered: wave 0 issues next-buffer TDM
// loads, waits s_wait_tensorcnt<=2 (in-order => current buffer complete),
// then the workgroup barrier releases compute. Fragments are contiguous
// ds_load_b128 pairs matching the ISA 16-bit lane layouts.
// ---------------------------------------------------------------------------
#define BM 128
#define BN 128
#define BK 32

__global__ __launch_bounds__(256) void gemm_bf16(
    const u16* __restrict__ A,    // [M,K] row-major
    const u16* __restrict__ BT,   // [N,K] row-major (B transposed)
    float* __restrict__ C, int M, int N, int K)
{
  __shared__ u16 As[2][BM * BK];   // 2 x 8 KB
  __shared__ u16 Bs[2][BN * BK];   // 2 x 8 KB
  const int t    = threadIdx.x;
  const int wave = t >> 5;
  const int lane = t & 31;
  const int m0 = blockIdx.y * BM;
  const int n0 = blockIdx.x * BN;

  f32x8 acc[8];
  #pragma unroll
  for (int i = 0; i < 8; ++i)
    #pragma unroll
    for (int r = 0; r < 8; ++r) acc[i][r] = 0.0f;

  const int hi    = (lane < 16) ? 0 : 1;
  const int rA    = wave * 16 + (lane & 15);   // A-frag row
  const int aLo   = hi ? 8 : 0;                // A-frag K: aLo..+7, aLo+16..+23
  const int nB    = lane & 15;                 // B-frag column
  const int bBase = hi ? 8 : 0;                // B-frag K: bBase..+7, bBase+16..+23

  if (wave == 0) {
    tdm_load_tile_128x32(A  + (size_t)m0 * K, (unsigned)(uintptr_t)&As[0][0], K);
    tdm_load_tile_128x32(BT + (size_t)n0 * K, (unsigned)(uintptr_t)&Bs[0][0], K);
  }

  int cur = 0;
  for (int k0 = 0; k0 < K; k0 += BK) {
    if (wave == 0) {
      if (k0 + BK < K) {
        tdm_load_tile_128x32(A  + (size_t)m0 * K + (k0 + BK),
                             (unsigned)(uintptr_t)&As[cur ^ 1][0], K);
        tdm_load_tile_128x32(BT + (size_t)n0 * K + (k0 + BK),
                             (unsigned)(uintptr_t)&Bs[cur ^ 1][0], K);
        __builtin_amdgcn_s_wait_tensorcnt(2);   // current buffer done (in-order)
      } else {
        __builtin_amdgcn_s_wait_tensorcnt(0);
      }
    }
    __syncthreads();

    const u16* as = &As[cur][0];
    const u16* bs = &Bs[cur][0];
    FragBF a;
    a.h[0] = *(const s16x8*)&as[rA * BK + aLo];
    a.h[1] = *(const s16x8*)&as[rA * BK + aLo + 16];
    FragBF bfr[8];
    #pragma unroll
    for (int s = 0; s < 8; ++s) {               // issue all B-frag LDS reads first
      const u16* bp = &bs[(s * 16 + nB) * BK + bBase];
      bfr[s].h[0] = *(const s16x8*)bp;
      bfr[s].h[1] = *(const s16x8*)(bp + 16);
    }
    #pragma unroll
    for (int s = 0; s < 8; ++s)
      acc[s] = __builtin_amdgcn_wmma_f32_16x16x32_bf16(
          false, a.bf, false, bfr[s].bf, (short)0, acc[s], false, false);
    __syncthreads();
    cur ^= 1;
  }

  const int colBase = n0 + nB;
  const int rowBase = m0 + wave * 16 + (hi ? 8 : 0);
  #pragma unroll
  for (int s = 0; s < 8; ++s)
    #pragma unroll
    for (int r = 0; r < 8; ++r)
      C[(size_t)(rowBase + r) * N + colBase + s * 16] = acc[s][r];
}

// ---------------------------------------------------------------------------
// inputs [B,S,N,1] -> h0 [B*N, 64] with h0[(b,n), s] = inputs[b,s,n,0]
// ---------------------------------------------------------------------------
__global__ void pack_h0(const float* __restrict__ in, float* __restrict__ h0) {
  int tid = blockIdx.x * 256 + threadIdx.x;
  if (tid >= B_ * S_ * N_) return;
  int n = tid % N_;
  int s = (tid / N_) % S_;
  int b = tid / (N_ * S_);
  h0[((size_t)(b * N_ + n)) * 64 + s] = in[tid];
}

// Y[M,Nc] = X[M,64] @ W[64,Nc] (+bias) (+X residual if addRes && Nc==64)
__global__ __launch_bounds__(256) void lin_f32(
    const float* __restrict__ X, const float* __restrict__ W,
    const float* __restrict__ bias, float* __restrict__ Y,
    int Nc, int addRes)
{
  __shared__ float sW[64 * 64];
  __shared__ float sX[4][64];
  const int t = threadIdx.x;
  const int c0 = blockIdx.x * 64;
  const int r = t >> 6, d = t & 63;
  const int row0 = blockIdx.y * 4;
  for (int i = t; i < 4096; i += 256) {
    int k = i >> 6, c = i & 63;
    sW[i] = W[k * Nc + c0 + c];
  }
  sX[r][d] = X[(size_t)(row0 + r) * 64 + d];
  __syncthreads();
  float acc = bias ? bias[c0 + d] : 0.0f;
  #pragma unroll 8
  for (int k = 0; k < 64; ++k) acc += sX[r][k] * sW[k * 64 + d];
  if (addRes) acc += sX[r][d];
  Y[(size_t)(row0 + r) * Nc + c0 + d] = acc;
}

__global__ void cvt_bf(const float* __restrict__ x, u16* __restrict__ y, int n) {
  int tid = blockIdx.x * 256 + threadIdx.x;
  if (tid < n) y[tid] = f2bf(x[tid]);
}

__global__ void fillk(float* __restrict__ p, float v, int n) {
  int tid = blockIdx.x * 256 + threadIdx.x;
  if (tid < n) p[tid] = v;
}

// per-row: A[m][n] = bf16( softmax_n( relu(R[m][n]) ) )
__global__ __launch_bounds__(256) void relu_softmax_row(
    const float* __restrict__ R, u16* __restrict__ Abf)
{
  __shared__ float red[256];
  const int row = blockIdx.x, t = threadIdx.x;
  const float* r = R + (size_t)row * N_;
  float mx = 0.0f;                                   // relu ensures max >= 0
  for (int c = t; c < N_; c += 256) mx = fmaxf(mx, fmaxf(r[c], 0.0f));
  red[t] = mx; __syncthreads();
  for (int s = 128; s > 0; s >>= 1) { if (t < s) red[t] = fmaxf(red[t], red[t + s]); __syncthreads(); }
  mx = red[0]; __syncthreads();
  float sum = 0.0f;
  for (int c = t; c < N_; c += 256) sum += __expf(fmaxf(r[c], 0.0f) - mx);
  red[t] = sum; __syncthreads();
  for (int s = 128; s > 0; s >>= 1) { if (t < s) red[t] += red[t + s]; __syncthreads(); }
  const float inv = 1.0f / red[0];
  for (int c = t; c < N_; c += 256)
    Abf[(size_t)row * N_ + c] = f2bf(__expf(fmaxf(r[c], 0.0f) - mx) * inv);
}

// h [B*N,64] f32 -> HpT [B*64][N] bf16 : HpT[(b*64+d)][n] = h[(b,n),d]
// (pre-transposed so the GEMM B-tile is a straight 2D copy for the TDM)
__global__ __launch_bounds__(256) void pack_hpT(const float* __restrict__ h,
                                                u16* __restrict__ HpT) {
  __shared__ u16 tile[64][65];
  const int n0 = blockIdx.x * 64;
  const int b  = blockIdx.y;
  const int t  = threadIdx.x;
  for (int i = t; i < 4096; i += 256) {
    int nl = i >> 6, d = i & 63;
    tile[nl][d] = f2bf(h[((size_t)(b * N_ + n0 + nl)) * 64 + d]);
  }
  __syncthreads();
  for (int i = t; i < 4096; i += 256) {
    int d = i >> 6, nl = i & 63;
    HpT[(size_t)(b * 64 + d) * N_ + n0 + nl] = tile[nl][d];
  }
}

// tgt [64][N] f32 -> tgtT [N][64] bf16
__global__ __launch_bounds__(256) void pack_tgtT(const float* __restrict__ tg,
                                                 u16* __restrict__ tgT) {
  __shared__ u16 tile[64][65];
  const int n0 = blockIdx.x * 64;
  const int t  = threadIdx.x;
  for (int i = t; i < 4096; i += 256) {
    int k = i >> 6, nl = i & 63;
    tile[nl][k] = f2bf(tg[(size_t)k * N_ + n0 + nl]);
  }
  __syncthreads();
  for (int i = t; i < 4096; i += 256) {
    int nl = i >> 6, k = i & 63;
    tgT[(size_t)(n0 + nl) * 64 + k] = tile[nl][k];
  }
}

// als/ald [bm, h] = xp[bm,h,:] . a{src,dst}[h,:]
__global__ void attn_coef(const float* __restrict__ xp,
                          const float* __restrict__ asrc, const float* __restrict__ adst,
                          float* __restrict__ als, float* __restrict__ ald, int H)
{
  int tid = blockIdx.x * 256 + threadIdx.x;
  if (tid >= BN_ * H) return;
  int hh = tid % H, bm = tid / H;
  const float* xr = xp + ((size_t)bm * H + hh) * 64;
  float s1 = 0.0f, s2 = 0.0f;
  #pragma unroll 8
  for (int k = 0; k < 64; ++k) {
    float v = xr[k];
    s1 += v * asrc[hh * 64 + k];
    s2 += v * adst[hh * 64 + k];
  }
  als[tid] = s1; ald[tid] = s2;
}

__global__ void gat_edge1(const float* __restrict__ als, const float* __restrict__ ald,
                          const int* __restrict__ eidx, float* __restrict__ ebuf,
                          float* __restrict__ mbuf, int H)
{
  int tid = blockIdx.x * 256 + threadIdx.x;
  if (tid >= B_ * ET_ * H) return;
  int hh = tid % H; int rest = tid / H;
  int e = rest % ET_; int b = rest / ET_;
  int s, d;
  if (e < E_) { s = eidx[e]; d = eidx[E_ + e]; } else { s = d = e - E_; }
  float v = als[(size_t)(b * N_ + s) * H + hh] + ald[(size_t)(b * N_ + d) * H + hh];
  v = (v > 0.0f) ? v : 0.2f * v;                     // leaky_relu 0.2
  ebuf[tid] = v;
  atomicMaxF(&mbuf[(size_t)(b * N_ + d) * H + hh], v);
}

__global__ void gat_edge2(float* __restrict__ ebuf, const float* __restrict__ mbuf,
                          float* __restrict__ zbuf, const int* __restrict__ eidx, int H)
{
  int tid = blockIdx.x * 256 + threadIdx.x;
  if (tid >= B_ * ET_ * H) return;
  int hh = tid % H; int rest = tid / H;
  int e = rest % ET_; int b = rest / ET_;
  int d = (e < E_) ? eidx[E_ + e] : (e - E_);
  size_t nd = (size_t)(b * N_ + d) * H + hh;
  float w = __expf(ebuf[tid] - mbuf[nd]);
  ebuf[tid] = w;
  atomicAdd(&zbuf[nd], w);
}

__global__ void gat_edge3(const float* __restrict__ ebuf, const float* __restrict__ zbuf,
                          const float* __restrict__ xp, const int* __restrict__ eidx,
                          float* __restrict__ gout, int H)
{
  int tid = blockIdx.x * 256 + threadIdx.x;
  if (tid >= B_ * ET_ * H * 64) return;
  int dch = tid & 63; int rest = tid >> 6;
  int hh = rest % H; rest /= H;
  int e = rest % ET_; int b = rest / ET_;
  int s, d;
  if (e < E_) { s = eidx[e]; d = eidx[E_ + e]; } else { s = d = e - E_; }
  float alpha = ebuf[((size_t)b * ET_ + e) * H + hh] /
                zbuf[(size_t)(b * N_ + d) * H + hh];
  float val = xp[((size_t)(b * N_ + s) * H + hh) * 64 + dch] * alpha;
  atomicAdd(&gout[((size_t)(b * N_ + d) * H + hh) * 64 + dch], val);
}

__global__ void gat_mean(const float* __restrict__ gout, const float* __restrict__ bg,
                         float* __restrict__ hcur, int H)
{
  int tid = blockIdx.x * 256 + threadIdx.x;
  if (tid >= BN_ * 64) return;
  int d = tid & 63, bm = tid >> 6;
  float s = 0.0f;
  for (int hh = 0; hh < H; ++hh) s += gout[((size_t)bm * H + hh) * 64 + d];
  hcur[tid] = s / (float)H + bg[d];
}

// g = sigmoid(adp_row @ Wl + bl); gating per mode. adp read from C [m, b*64+d].
__global__ __launch_bounds__(256) void gate_kernel(
    const float* __restrict__ C, const float* __restrict__ Wl, const float* __restrict__ bl,
    const float* __restrict__ Wo, const float* __restrict__ bo,
    const float* __restrict__ h, const float* __restrict__ hcur,
    float* __restrict__ out, int mode)
{
  __shared__ float sWl[64 * 64];
  __shared__ float sWo[64 * 64];
  __shared__ float sAdp[4][64];
  __shared__ float sH[4][64];
  const int t = threadIdx.x;
  for (int i = t; i < 4096; i += 256) sWl[i] = Wl[i];
  if (mode == 0) for (int i = t; i < 4096; i += 256) sWo[i] = Wo[i];
  const int r = t >> 6, d = t & 63;
  const int bm = blockIdx.x * 4 + r;
  const int b = bm >> 12;                 // / N_
  const int m = bm & (N_ - 1);
  sAdp[r][d] = C[(size_t)m * NCOL_ + b * 64 + d];
  sH[r][d]   = h[(size_t)bm * 64 + d];
  __syncthreads();
  float acc = bl[d];
  #pragma unroll 8
  for (int k = 0; k < 64; ++k) acc += sAdp[r][k] * sWl[k * 64 + d];
  const float g = 1.0f / (1.0f + __expf(-acc));
  const float hv = sH[r][d];
  float res;
  if (mode == 0) {
    float o = bo[d];
    #pragma unroll 8
    for (int k = 0; k < 64; ++k) o += sH[r][k] * sWo[k * 64 + d];
    res = tanhf(hv) * g + o * (1.0f - g);
  } else {
    float hc = hcur[(size_t)bm * 64 + d];
    float act = (mode == 1) ? (hc > 0.0f ? hc : 0.01f * hc) : fmaxf(hc, 0.0f);
    res = act * g + hv * (1.0f - g);
  }
  out[(size_t)bm * 64 + d] = res;
}

#define GRID1(n) dim3((unsigned)(((n) + 255) / 256))

extern "C" void kernel_launch(void* const* d_in, const int* in_sizes, int n_in,
                              void* d_out, int out_size, void* d_ws, size_t ws_size,
                              hipStream_t stream)
{
  (void)in_sizes; (void)n_in; (void)out_size; (void)ws_size;
  const float* inputs = (const float*)d_in[0];
  const int*   eidx   = (const int*)d_in[1];
  const float* W_seq  = (const float*)d_in[2];
  const float* b_seq  = (const float*)d_in[3];
  const float* src_e  = (const float*)d_in[4];
  const float* tgt_e  = (const float*)d_in[5];
  const float* Wg[3]   = {(const float*)d_in[6],  (const float*)d_in[12], (const float*)d_in[18]};
  const float* asrc[3] = {(const float*)d_in[7],  (const float*)d_in[13], (const float*)d_in[19]};
  const float* adst[3] = {(const float*)d_in[8],  (const float*)d_in[14], (const float*)d_in[20]};
  const float* bg[3]   = {(const float*)d_in[9],  (const float*)d_in[15], (const float*)d_in[21]};
  const float* Wl[3]   = {(const float*)d_in[10], (const float*)d_in[16], (const float*)d_in[22]};
  const float* bl[3]   = {(const float*)d_in[11], (const float*)d_in[17], (const float*)d_in[23]};
  const float* Wo = (const float*)d_in[24];
  const float* bo = (const float*)d_in[25];

  char* wsb = (char*)d_ws;
  size_t off = 0;
  auto alloc = [&](size_t bytes) -> char* {
    char* p = wsb + off;
    off += (bytes + 255) & ~(size_t)255;
    return p;
  };
  const size_t SZ_H = (size_t)BN_ * 64 * sizeof(float);
  float* h0   = (float*)alloc(SZ_H);
  float* hA   = (float*)alloc(SZ_H);
  float* hB   = (float*)alloc(SZ_H);
  float* hcur = (float*)alloc(SZ_H);
  u16* srcbf  = (u16*)alloc((size_t)N_ * 64 * 2);
  u16* tgtT   = (u16*)alloc((size_t)N_ * 64 * 2);       // [N][64]
  u16* Abf    = (u16*)alloc((size_t)N_ * N_ * 2);       // [M][K]
  u16* HpT    = (u16*)alloc((size_t)NCOL_ * N_ * 2);    // [1024][4096]
  float* Cbuf = (float*)alloc((size_t)N_ * NCOL_ * sizeof(float));
  float* als  = (float*)alloc((size_t)BN_ * 3 * sizeof(float));
  float* ald  = (float*)alloc((size_t)BN_ * 3 * sizeof(float));
  float* mbuf = (float*)alloc((size_t)BN_ * 3 * sizeof(float));
  float* zbuf = (float*)alloc((size_t)BN_ * 3 * sizeof(float));
  float* ebuf = (float*)alloc((size_t)B_ * ET_ * 3 * sizeof(float));
  // big overlap region: R (64 MB) is dead before xp/gout (48 MB each) go live
  const size_t SZ_XP = (size_t)BN_ * 192 * sizeof(float);
  char* region = alloc(2 * SZ_XP);
  float* Rbuf = (float*)region;
  float* xp   = (float*)region;
  float* gout = (float*)(region + SZ_XP);

  // ---- setup: h = (inputs permuted) @ W_seq + b_seq + residual
  pack_h0<<<GRID1(B_ * S_ * N_), 256, 0, stream>>>(inputs, h0);
  lin_f32<<<dim3(1, BN_ / 4), 256, 0, stream>>>(h0, W_seq, b_seq, hA, 64, 1);

  // ---- A = softmax(relu(src_embed @ tgt_embed)), stored bf16 [M][K]
  cvt_bf<<<GRID1(N_ * 64), 256, 0, stream>>>(src_e, srcbf, N_ * 64);
  pack_tgtT<<<dim3(N_ / 64), 256, 0, stream>>>(tgt_e, tgtT);
  gemm_bf16<<<dim3(N_ / BN, N_ / BM), 256, 0, stream>>>(srcbf, tgtT, Rbuf, N_, N_, 64);
  relu_softmax_row<<<N_, 256, 0, stream>>>(Rbuf, Abf);

  float* cur = hA;
  float* nxt = hB;
  for (int i = 0; i < 3; ++i) {
    const int H = (i == 2) ? 1 : 3;
    if (i > 0) {
      lin_f32<<<dim3(H, BN_ / 4), 256, 0, stream>>>(cur, Wg[i], nullptr, xp, H * 64, 0);
      attn_coef<<<GRID1(BN_ * H), 256, 0, stream>>>(xp, asrc[i], adst[i], als, ald, H);
      fillk<<<GRID1(BN_ * H), 256, 0, stream>>>(mbuf, -INFINITY, BN_ * H);
      fillk<<<GRID1(BN_ * H), 256, 0, stream>>>(zbuf, 0.0f, BN_ * H);
      fillk<<<GRID1(BN_ * H * 64), 256, 0, stream>>>(gout, 0.0f, BN_ * H * 64);
      gat_edge1<<<GRID1(B_ * ET_ * H), 256, 0, stream>>>(als, ald, eidx, ebuf, mbuf, H);
      gat_edge2<<<GRID1(B_ * ET_ * H), 256, 0, stream>>>(ebuf, mbuf, zbuf, eidx, H);
      gat_edge3<<<GRID1(B_ * ET_ * H * 64), 256, 0, stream>>>(ebuf, zbuf, xp, eidx, gout, H);
      gat_mean<<<GRID1(BN_ * 64), 256, 0, stream>>>(gout, bg[i], hcur, H);
    }
    // adp: fused-batch GEMM  C[4096,1024] = A_bf16 @ HpT^T  (TDM-streamed)
    pack_hpT<<<dim3(N_ / 64, B_), 256, 0, stream>>>(cur, HpT);
    gemm_bf16<<<dim3(NCOL_ / BN, N_ / BM), 256, 0, stream>>>(Abf, HpT, Cbuf, N_, NCOL_, N_);
    float* dst = (i == 2) ? (float*)d_out : nxt;
    gate_kernel<<<BN_ / 4, 256, 0, stream>>>(Cbuf, Wl[i], bl[i], Wo, bo, cur, hcur, dst, i);
    if (i < 2) { float* tmp = cur; cur = nxt; nxt = tmp; }
  }
}
